// AxialBlock_12764642803742
// MI455X (gfx1250) — matrix operand order repaired and checked by reading, not compiled, not run on hardware
//
#include <hip/hip_runtime.h>
#include <stdint.h>

// ---------- CDNA5 WMMA types ----------
typedef __attribute__((ext_vector_type(16))) __bf16   v16bf;
typedef __attribute__((ext_vector_type(8)))  float    v8f;
typedef __attribute__((ext_vector_type(4)))  unsigned u32x4;

union Frag16 {                 // 16 bf16 = one WMMA A/B fragment (8 VGPRs)
    v16bf          v;
    u32x4          q[2];
    unsigned short u[16];
};

static __device__ __forceinline__ unsigned short f2bf(float f) {
    union { float f; unsigned u; } a; a.f = f;
    unsigned u = a.u;
    u += 0x7fffu + ((u >> 16) & 1u);       // round-to-nearest-even
    return (unsigned short)(u >> 16);
}

static __device__ __forceinline__ v8f wmma_bf16(const Frag16& a, const Frag16& b, v8f c) {
    // (neg_a, A, neg_b, B, c_mod, C, reuse_a, reuse_b)
    return __builtin_amdgcn_wmma_f32_16x16x32_bf16(false, a.v, false, b.v,
                                                   (short)0, c, false, false);
}

// Problem constants (from setup_inputs): B=4, C=512, T=16, H=32, W=32
#define NTOK   65536          // B * T*H*W
#define SPB    16384          // T*H*W (tokens per batch)
#define CCH    512

// ---------- pack x: (B,C,S) f32 -> token-major bf16 Xt[(b*S+s)*512 + c] ----------
__global__ __launch_bounds__(256) void pack_x(const float* __restrict__ x,
                                              unsigned short* __restrict__ Xt) {
    size_t tid = (size_t)blockIdx.x * 256 + threadIdx.x;     // (b*512 + c)*16384 + s
    size_t s  = tid & 16383;
    size_t bc = tid >> 14;
    size_t c  = bc & 511;
    size_t b  = bc >> 9;
    Xt[((b << 14) + s) * CCH + c] = f2bf(x[tid]);
}

// ---------- pack weights: f32 -> bf16 (same row-major layout) ----------
__global__ __launch_bounds__(256) void pack_w(const float* __restrict__ in,
                                              unsigned short* __restrict__ out, int n) {
    int tid = blockIdx.x * 256 + threadIdx.x;
    if (tid < n) out[tid] = f2bf(in[tid]);
}

// ====================================================================================
// Projection GEMM family: D[g, co] = sum_c W[co, c] * Src_t[g, c]
// grid = (512/64, 65536/128), block = 128 (4 waves)
// wave tile = 32(co) x 64(tok) = 2x4 WMMA tiles, K double-buffered in 32-element steps
// ====================================================================================
struct GemmCtx {
    const unsigned short* wr0;   // W row for m-frag 0
    const unsigned short* wr1;   // W row for m-frag 1
    const unsigned short* xr0;   // Src rows for n-frags 0..3
    const unsigned short* xr1;
    const unsigned short* xr2;
    const unsigned short* xr3;
    int koffA, koffB;
};

static __device__ __forceinline__ void load_a(const GemmCtx& c, Frag16 a[2], int kk) {
    a[0].q[0] = *(const u32x4*)(c.wr0 + kk + c.koffA);
    a[0].q[1] = *(const u32x4*)(c.wr0 + kk + 8 + c.koffA);
    a[1].q[0] = *(const u32x4*)(c.wr1 + kk + c.koffA);
    a[1].q[1] = *(const u32x4*)(c.wr1 + kk + 8 + c.koffA);
}
static __device__ __forceinline__ void load_b(const GemmCtx& c, Frag16 b[4], int kk) {
    b[0].q[0] = *(const u32x4*)(c.xr0 + kk + c.koffB);
    b[0].q[1] = *(const u32x4*)(c.xr0 + kk + c.koffB + 8);
    b[1].q[0] = *(const u32x4*)(c.xr1 + kk + c.koffB);
    b[1].q[1] = *(const u32x4*)(c.xr1 + kk + c.koffB + 8);
    b[2].q[0] = *(const u32x4*)(c.xr2 + kk + c.koffB);
    b[2].q[1] = *(const u32x4*)(c.xr2 + kk + c.koffB + 8);
    b[3].q[0] = *(const u32x4*)(c.xr3 + kk + c.koffB);
    b[3].q[1] = *(const u32x4*)(c.xr3 + kk + c.koffB + 8);
}

// Shared mainloop: double-buffered K, 8 WMMAs per 32-K step group
static __device__ __forceinline__ void gemm_mainloop(const GemmCtx& c, v8f acc[2][4]) {
    Frag16 a0[2], b0[4], a1[2], b1[4];
    load_a(c, a0, 0);
    load_b(c, b0, 0);
    for (int kk = 0; kk < CCH; kk += 64) {
        // issue loads for kk+32 before consuming kk's fragments
        load_a(c, a1, kk + 32);
        load_b(c, b1, kk + 32);
        const int kp = (kk + 128) & 511;               // prefetch 2 panels ahead
        // locality 3 -> WGP-scope prefetch: pulls into all cache levels (near cache),
        // which is what matters since the panels are L2-resident already.
        __builtin_prefetch(c.wr0 + kp, 0, 3);
        __builtin_prefetch(c.wr1 + kp, 0, 3);
        __builtin_prefetch(c.xr0 + kp, 0, 3);
        __builtin_prefetch(c.xr1 + kp, 0, 3);
        __builtin_prefetch(c.xr2 + kp, 0, 3);
        __builtin_prefetch(c.xr3 + kp, 0, 3);
#pragma unroll
        for (int mi = 0; mi < 2; ++mi)
#pragma unroll
            for (int ni = 0; ni < 4; ++ni)
                acc[mi][ni] = wmma_bf16(a0[mi], b0[ni], acc[mi][ni]);
        // loads for kk+64 (wraps harmlessly on last iter; results unused)
        const int kn = (kk + 64) & 511;
        load_a(c, a0, kn);
        load_b(c, b0, kn);
#pragma unroll
        for (int mi = 0; mi < 2; ++mi)
#pragma unroll
            for (int ni = 0; ni < 4; ++ni)
                acc[mi][ni] = wmma_bf16(a1[mi], b1[ni], acc[mi][ni]);
    }
}

static __device__ __forceinline__ void gemm_setup(GemmCtx& c,
                                                  const unsigned short* Wbf,
                                                  const unsigned short* Src,
                                                  int& co_base, int& g_base,
                                                  int& half, int& nl) {
    const int lane = threadIdx.x & 31;
    const int wave = threadIdx.x >> 5;
    half = lane >> 4;
    nl   = lane & 15;
    co_base = blockIdx.x * 64 + (wave >> 1) * 32;
    g_base  = blockIdx.y * 128 + (wave & 1) * 64;
    c.koffA = half * 16;    // A: lanes 0-15 K 0..15; 16-31 K 16..31
    c.koffB = half * 16;    // B: lanes 0-15 K 0..15; 16-31 K 16..31
    c.wr0 = Wbf + (size_t)(co_base + nl) * CCH;
    c.wr1 = Wbf + (size_t)(co_base + 16 + nl) * CCH;
    c.xr0 = Src + (size_t)(g_base + nl) * CCH;
    c.xr1 = Src + (size_t)(g_base + 16 + nl) * CCH;
    c.xr2 = Src + (size_t)(g_base + 32 + nl) * CCH;
    c.xr3 = Src + (size_t)(g_base + 48 + nl) * CCH;
}

// ---------- Q/K/V projection: bf16 token-major output -----------------------------
__global__ __launch_bounds__(128) void gemm_qkv(const unsigned short* __restrict__ Wbf,
                                                const unsigned short* __restrict__ Xt,
                                                unsigned short* __restrict__ Out) {
    GemmCtx c; int co_base, g_base, half, nl;
    gemm_setup(c, Wbf, Xt, co_base, g_base, half, nl);
    v8f acc[2][4] = {};
    gemm_mainloop(c, acc);
    // D layout: VGPR r -> row (r + 8*half); col = nl.  8 contiguous co per lane.
#pragma unroll
    for (int mi = 0; mi < 2; ++mi)
#pragma unroll
        for (int ni = 0; ni < 4; ++ni) {
            union { unsigned short u[8]; u32x4 q; } pk;
#pragma unroll
            for (int r = 0; r < 8; ++r) pk.u[r] = f2bf(acc[mi][ni][r]);
            const int g = g_base + ni * 16 + nl;
            *(u32x4*)(Out + (size_t)g * CCH + co_base + mi * 16 + half * 8) = pk.q;
        }
}

// ---------- output projection: out[b,co,s] (+)= Wo @ O + bo (f32) ------------------
__global__ __launch_bounds__(128) void gemm_out(const unsigned short* __restrict__ Wbf,
                                                const unsigned short* __restrict__ Ot,
                                                const float* __restrict__ bias,
                                                float* __restrict__ out, int accumulate) {
    GemmCtx c; int co_base, g_base, half, nl;
    gemm_setup(c, Wbf, Ot, co_base, g_base, half, nl);
    v8f acc[2][4] = {};
    gemm_mainloop(c, acc);
#pragma unroll
    for (int mi = 0; mi < 2; ++mi)
#pragma unroll
        for (int ni = 0; ni < 4; ++ni) {
            const int g = g_base + ni * 16 + nl;
            const int b = g >> 14;
            const int s = g & 16383;
#pragma unroll
            for (int r = 0; r < 8; ++r) {
                const int co = co_base + mi * 16 + 8 * half + r;
                size_t idx = ((size_t)b * CCH + co) * SPB + s;
                float v = acc[mi][ni][r] + bias[co];
                if (accumulate) v += out[idx];
                out[idx] = v;
            }
        }
}

// ---------- axial attention: one slice (b, head, outer) per wave -----------------
// token s_j = b*SPB + base(o) + j*stride,  base = (o>>divshift)*mulq + (o&mask)*mulr
// scores = WMMA (K=d=64, 2 steps); softmax via wave32 shuffles; O = P @ V^T via WMMA
// with K zero-padded to 32 so L=16 reuses 16x16x32. O overwrites Q in place (disjoint
// (token, head-channel) region per slice).
template <int L>
__global__ __launch_bounds__(128) void attn_axis(const unsigned short* __restrict__ Qt,
                                                 const unsigned short* __restrict__ Kt,
                                                 const unsigned short* __restrict__ Vt,
                                                 unsigned short* __restrict__ Ot,
                                                 int stride, int n_outer,
                                                 int divshift, int mulq, int mulr) {
    constexpr int nT = L / 16;
    __shared__ __align__(16) unsigned short sV[4][64 * 32];  // V^T, cols zero-padded
    __shared__ __align__(16) unsigned short sP[4][32 * 40];  // probs bf16, stride 40

    const int lane = threadIdx.x & 31;
    const int wave = threadIdx.x >> 5;
    const int half = lane >> 4;
    const int nl   = lane & 15;

    const int slice = blockIdx.x * 4 + wave;
    const int per_b = 8 * n_outer;
    const int b     = slice / per_b;
    const int rem   = slice - b * per_b;
    const int head  = rem / n_outer;
    const int o     = rem - head * n_outer;
    const int sbase = (o >> divshift) * mulq + (o & ((1 << divshift) - 1)) * mulr;
    const size_t tokbase = (size_t)b * SPB + sbase;
    const int cbase = head * 64;

    // stage V^T into LDS: sV[dd*32 + j], j >= L zero-padded
    unsigned short* vbuf = sV[wave];
    for (int it = 0; it < 8; ++it) {
        int c  = it * 32 + lane;            // c in [0,256): j = c>>3, chunk = (c&7)*8
        int j  = c >> 3;
        int k8 = (c & 7) * 8;
        if (j < L) {
            union { u32x4 q; unsigned short u[8]; } w;
            w.q = *(const u32x4*)(Vt + (tokbase + (size_t)j * stride) * CCH + cbase + k8);
#pragma unroll
            for (int t = 0; t < 8; ++t) vbuf[(k8 + t) * 32 + j] = w.u[t];
        } else {
#pragma unroll
            for (int t = 0; t < 8; ++t) vbuf[(k8 + t) * 32 + j] = 0;
        }
    }

    // scores S = Q K^T (M=L, N=L, K=64)
    v8f accS[nT][nT] = {};
#pragma unroll
    for (int mt = 0; mt < nT; ++mt) {
#pragma unroll
        for (int kk = 0; kk < 64; kk += 32) {
            Frag16 a;
            const unsigned short* qrow =
                Qt + (tokbase + (size_t)(mt * 16 + nl) * stride) * CCH + cbase + kk;
            a.q[0] = *(const u32x4*)(qrow + half * 16);
            a.q[1] = *(const u32x4*)(qrow + 8 + half * 16);
#pragma unroll
            for (int nt = 0; nt < nT; ++nt) {
                Frag16 bf;
                const unsigned short* krow =
                    Kt + (tokbase + (size_t)(nt * 16 + nl) * stride) * CCH + cbase + kk + half * 16;
                bf.q[0] = *(const u32x4*)(krow);
                bf.q[1] = *(const u32x4*)(krow + 8);
                accS[mt][nt] = wmma_bf16(a, bf, accS[mt][nt]);
            }
        }
    }

    // softmax over j: each row lives in one 16-lane half of the wave
    const float scale = 0.125f;   // 1/sqrt(64)
    unsigned short* pbuf = sP[wave];
#pragma unroll
    for (int mt = 0; mt < nT; ++mt) {
#pragma unroll
        for (int r = 0; r < 8; ++r) {
            float v0 = accS[mt][0][r] * scale;
            float v1 = (nT > 1) ? accS[mt][nT - 1][r] * scale : v0;
            float mx = fmaxf(v0, v1);
#pragma unroll
            for (int msk = 1; msk < 16; msk <<= 1) mx = fmaxf(mx, __shfl_xor(mx, msk, 32));
            float e0 = __expf(v0 - mx);
            float e1 = (nT > 1) ? __expf(v1 - mx) : 0.0f;
            float sm = e0 + e1;
#pragma unroll
            for (int msk = 1; msk < 16; msk <<= 1) sm += __shfl_xor(sm, msk, 32);
            float inv = 1.0f / sm;
            const int mrow = mt * 16 + r + 8 * half;
            pbuf[mrow * 40 + nl]      = f2bf(e0 * inv);
            pbuf[mrow * 40 + 16 + nl] = (nT > 1) ? f2bf(e1 * inv) : (unsigned short)0;
        }
    }
    __syncthreads();   // make LDS V^T and probs visible across lanes

    // O = P (Lx32 padded) @ V^T (32x64): M=L, N=64, K=32, one WMMA K-step
    v8f accO[nT][4] = {};
#pragma unroll
    for (int mt = 0; mt < nT; ++mt) {
        Frag16 a;
        const unsigned short* prow = pbuf + (mt * 16 + nl) * 40 + half * 16;
        a.q[0] = *(const u32x4*)(prow);
        a.q[1] = *(const u32x4*)(prow + 8);
#pragma unroll
        for (int nt = 0; nt < 4; ++nt) {
            Frag16 bf;
            const unsigned short* vrow = vbuf + (nt * 16 + nl) * 32 + half * 16;
            bf.q[0] = *(const u32x4*)(vrow);
            bf.q[1] = *(const u32x4*)(vrow + 8);
            accO[mt][nt] = wmma_bf16(a, bf, accO[mt][nt]);
        }
    }

    // write O token-major bf16 (overwrites Q region of this slice only)
#pragma unroll
    for (int mt = 0; mt < nT; ++mt) {
#pragma unroll
        for (int nt = 0; nt < 4; ++nt) {
#pragma unroll
            for (int r = 0; r < 8; ++r) {
                const int i  = mt * 16 + r + 8 * half;
                const int dd = nt * 16 + nl;
                Ot[(tokbase + (size_t)i * stride) * CCH + cbase + dd] = f2bf(accO[mt][nt][r]);
            }
        }
    }
}

// ------------------------------- host launcher ------------------------------------
extern "C" void kernel_launch(void* const* d_in, const int* in_sizes, int n_in,
                              void* d_out, int out_size, void* d_ws, size_t ws_size,
                              hipStream_t stream) {
    (void)in_sizes; (void)n_in; (void)out_size; (void)ws_size;
    const float* x   = (const float*)d_in[0];
    float*       out = (float*)d_out;

    // workspace layout (bytes): Xt 64MB | 12x W bf16 6MB | Qt 64MB | Kt 64MB | Vt 64MB
    char* ws = (char*)d_ws;
    unsigned short* Xt = (unsigned short*)(ws);
    unsigned short* Wb = (unsigned short*)(ws + 67108864);
    unsigned short* Qt = (unsigned short*)(ws + 67108864 + 6291456);
    unsigned short* Kt = Qt + (size_t)NTOK * CCH;
    unsigned short* Vt = Kt + (size_t)NTOK * CCH;

    pack_x<<<131072, 256, 0, stream>>>(x, Xt);

    const int widx[12] = {1, 2, 3, 4, 6, 7, 8, 9, 11, 12, 13, 14};
    for (int m = 0; m < 12; ++m)
        pack_w<<<1024, 256, 0, stream>>>((const float*)d_in[widx[m]],
                                         Wb + (size_t)m * 262144, 262144);

    dim3 gg(8, 512), gb(128);
    for (int a = 0; a < 3; ++a) {
        const unsigned short* wq = Wb + (size_t)(a * 4 + 0) * 262144;
        const unsigned short* wk = Wb + (size_t)(a * 4 + 1) * 262144;
        const unsigned short* wv = Wb + (size_t)(a * 4 + 2) * 262144;
        const unsigned short* wo = Wb + (size_t)(a * 4 + 3) * 262144;
        const float* bo = (const float*)d_in[1 + a * 5 + 4];

        gemm_qkv<<<gg, gb, 0, stream>>>(wq, Xt, Qt);
        gemm_qkv<<<gg, gb, 0, stream>>>(wk, Xt, Kt);
        gemm_qkv<<<gg, gb, 0, stream>>>(wv, Xt, Vt);

        if (a == 0)       // width: L=32, stride=1,    outer o = t*32+h -> base o*32
            attn_axis<32><<<4096, 128, 0, stream>>>(Qt, Kt, Vt, Qt, 1, 512, 0, 32, 0);
        else if (a == 1)  // height: L=32, stride=32,  o = t*32+w -> base (o>>5)*1024 + (o&31)
            attn_axis<32><<<4096, 128, 0, stream>>>(Qt, Kt, Vt, Qt, 32, 512, 5, 1024, 1);
        else              // time: L=16, stride=1024,  o = h*32+w -> base o
            attn_axis<16><<<8192, 128, 0, stream>>>(Qt, Kt, Vt, Qt, 1024, 1024, 0, 1, 0);

        gemm_out<<<gg, gb, 0, stream>>>(wo, Qt, bo, out, a > 0 ? 1 : 0);
    }
}
